// CausalAttention_V0_78391743087197
// MI455X (gfx1250) — compile-verified, matrix operand order checked by reading, not run on hardware
//
#include <hip/hip_runtime.h>
#include <stdint.h>

typedef __attribute__((ext_vector_type(16))) __bf16 v16bf;
typedef __attribute__((ext_vector_type(8)))  float  v8f;

union Frag16 { v16bf bf; uint4 q[2]; };

__device__ __forceinline__ unsigned short f2bf(float f) {
    union { float f; uint32_t u; } a; a.f = f;
    uint32_t u = a.u;
    uint32_t r = (u + 0x7FFFu + ((u >> 16) & 1u)) >> 16;
    return (unsigned short)r;
}

// Async global->LDS copy of 16 bytes per lane (CDNA5 GLOBAL_LOAD_ASYNC_TO_LDS_B128).
// ldsoff = LDS byte offset (low 32 bits of generic pointer to __shared__), gp = global addr.
__device__ __forceinline__ void async_cp16(uint32_t ldsoff, const void* gp) {
    asm volatile("global_load_async_to_lds_b128 %0, %1, off"
                 :: "v"(ldsoff), "v"(gp) : "memory");
}
__device__ __forceinline__ void async_wait0() {
    asm volatile("s_wait_asynccnt 0x0" ::: "memory");
}

// ---------------------------------------------------------------- convert
__global__ void cvt_f32_bf16(const float* __restrict__ in,
                             unsigned short* __restrict__ out, int n) {
    int i = blockIdx.x * blockDim.x + threadIdx.x;
    int stride = gridDim.x * blockDim.x;
    for (; i < n; i += stride) out[i] = f2bf(in[i]);
}

// ---------------------------------------------------------------- GEMM
// C(MxN) = A(MxK, bf16 row-major, lda) * B^T where B is (NxK, bf16 row-major, ldb)
// mode 0: C bf16 row-major [row*ldc+col]
// mode 1: C bf16 transposed [col*ldc+row]
// mode 2: C f32  row-major [row*ldc+col]
#define BM 64
#define BN 64
#define BK 32
#define LDT 40  // LDS row stride in bf16 (16B aligned, conflict-free)

__device__ __forceinline__ void stage_tile_async(
    const unsigned short* __restrict__ g, int ldg, int rbase, int k0,
    unsigned short* s, int tid)
{
    #pragma unroll
    for (int i = 0; i < 2; ++i) {
        int c   = tid + i * 128;       // 0..255
        int row = c >> 2;
        int col = (c & 3) << 3;
        uint32_t lds = (uint32_t)(uintptr_t)&s[row * LDT + col];
        async_cp16(lds, &g[(size_t)(rbase + row) * ldg + k0 + col]);
    }
}

__global__ __launch_bounds__(128) void gemm_bf16(
    const unsigned short* __restrict__ A, int lda,
    const unsigned short* __restrict__ B, int ldb,
    void* __restrict__ C, int K, int mode, int ldc)
{
    __shared__ unsigned short sA[2][BM * LDT];
    __shared__ unsigned short sB[2][BN * LDT];

    const int tid  = threadIdx.x;
    const int wv   = tid >> 5;
    const int lane = tid & 31;
    const int lrow = lane & 15;
    const int hi   = lane >> 4;

    const int mbase = blockIdx.x * BM;
    const int nbase = blockIdx.y * BN;

    v8f acc[4] = {};

    // prologue: stage K-step 0 into buffer 0
    stage_tile_async(A, lda, mbase, 0, sA[0], tid);
    stage_tile_async(B, ldb, nbase, 0, sB[0], tid);

    int buf = 0;
    for (int k0 = 0; k0 < K; k0 += BK) {
        async_wait0();
        __syncthreads();            // tile[buf] ready; everyone done reading tile[buf^1]

        if (k0 + BK < K) {          // overlap next tile's DMA with this tile's WMMAs
            stage_tile_async(A, lda, mbase, k0 + BK, sA[buf ^ 1], tid);
            stage_tile_async(B, ldb, nbase, k0 + BK, sB[buf ^ 1], tid);
        }

        Frag16 fa, fb[4];
        fa.q[0] = *(const uint4*)&sA[buf][(wv * 16 + lrow) * LDT + hi * 8];
        fa.q[1] = *(const uint4*)&sA[buf][(wv * 16 + lrow) * LDT + 16 + hi * 8];
        #pragma unroll
        for (int t = 0; t < 4; ++t) {
            fb[t].q[0] = *(const uint4*)&sB[buf][(t * 16 + lrow) * LDT + hi * 8];
            fb[t].q[1] = *(const uint4*)&sB[buf][(t * 16 + lrow) * LDT + 16 + hi * 8];
        }
        #pragma unroll
        for (int t = 0; t < 4; ++t)
            acc[t] = __builtin_amdgcn_wmma_f32_16x16x32_bf16(
                false, fa.bf, false, fb[t].bf, (short)0, acc[t], false, false);

        buf ^= 1;
    }

    const int orow = mbase + wv * 16 + hi * 8;
    #pragma unroll
    for (int t = 0; t < 4; ++t) {
        int col = nbase + t * 16 + lrow;
        #pragma unroll
        for (int r = 0; r < 8; ++r) {
            float v = acc[t][r];
            int row = orow + r;
            if (mode == 0)
                ((unsigned short*)C)[(size_t)row * ldc + col] = f2bf(v);
            else if (mode == 1)
                ((unsigned short*)C)[(size_t)col * ldc + row] = f2bf(v);
            else
                ((float*)C)[(size_t)row * ldc + col] = v;
        }
    }
}

// ---------------------------------------------------------------- softmax(QK^T)
#define QSTR 1032  // padded LDS stride for the 16x1024 Q strip

__device__ __forceinline__ v8f s_tile(const unsigned short* sQ,
                                      const unsigned short* __restrict__ Kb,
                                      int jt, int lrow, int hi)
{
    v8f acc = {};
    const unsigned short* krow = Kb + (size_t)(jt + lrow) * 1024;
    #pragma unroll 4
    for (int k0 = 0; k0 < 1024; k0 += 32) {
        Frag16 fa, fb;
        fa.q[0] = *(const uint4*)&sQ[lrow * QSTR + k0 + hi * 8];
        fa.q[1] = *(const uint4*)&sQ[lrow * QSTR + k0 + 16 + hi * 8];
        fb.q[0] = *(const uint4*)&krow[k0 + hi * 8];
        fb.q[1] = *(const uint4*)&krow[k0 + 16 + hi * 8];
        acc = __builtin_amdgcn_wmma_f32_16x16x32_bf16(
            false, fa.bf, false, fb.bf, (short)0, acc, false, false);
    }
    return acc;
}

__global__ __launch_bounds__(256) void attn_softmax(
    const unsigned short* __restrict__ Qb,
    const unsigned short* __restrict__ Kb,
    unsigned short* __restrict__ Pb)
{
    __shared__ unsigned short sQ[16 * QSTR];
    __shared__ float sM[8][16];
    __shared__ float sL[8][16];

    const int tid   = threadIdx.x;
    const int wv    = tid >> 5;
    const int lane  = tid & 31;
    const int lrow  = lane & 15;
    const int hi    = lane >> 4;
    const int ibase = blockIdx.x * 16;

    // stage 16x1024 bf16 Q strip via async DMA to LDS
    #pragma unroll
    for (int i = 0; i < 8; ++i) {
        int c   = tid + i * 256;       // 0..2047
        int row = c >> 7;
        int col = (c & 127) << 3;
        uint32_t lds = (uint32_t)(uintptr_t)&sQ[row * QSTR + col];
        async_cp16(lds, &Qb[(size_t)(ibase + row) * 1024 + col]);
    }
    async_wait0();
    __syncthreads();

    const float scale = 0.03125f;  // 1/sqrt(1024)
    float m_run[8], l_run[8];
    #pragma unroll
    for (int r = 0; r < 8; ++r) { m_run[r] = -3.0e38f; l_run[r] = 0.f; }

    const int j0 = wv * 512;

    // pass 1: online row max / sumexp over this wave's j-strip
    for (int jt = j0; jt < j0 + 512; jt += 16) {
        v8f s = s_tile(sQ, Kb, jt, lrow, hi);
        #pragma unroll
        for (int r = 0; r < 8; ++r) {
            float t  = s[r] * scale;
            float tm = t;
            tm = fmaxf(tm, __shfl_xor(tm, 1));
            tm = fmaxf(tm, __shfl_xor(tm, 2));
            tm = fmaxf(tm, __shfl_xor(tm, 4));
            tm = fmaxf(tm, __shfl_xor(tm, 8));
            float m_new = fmaxf(m_run[r], tm);
            float p  = __expf(t - m_new);
            p += __shfl_xor(p, 1);
            p += __shfl_xor(p, 2);
            p += __shfl_xor(p, 4);
            p += __shfl_xor(p, 8);
            l_run[r] = l_run[r] * __expf(m_run[r] - m_new) + p;
            m_run[r] = m_new;
        }
    }

    // combine the 8 waves' stats
    if (lrow == 0) {
        #pragma unroll
        for (int r = 0; r < 8; ++r) {
            sM[wv][hi * 8 + r] = m_run[r];
            sL[wv][hi * 8 + r] = l_run[r];
        }
    }
    __syncthreads();

    float m_fin[8], l_fin[8];
    #pragma unroll
    for (int r = 0; r < 8; ++r) {
        int row = hi * 8 + r;
        float m = -3.0e38f;
        #pragma unroll
        for (int w = 0; w < 8; ++w) m = fmaxf(m, sM[w][row]);
        float l = 0.f;
        #pragma unroll
        for (int w = 0; w < 8; ++w) l += sL[w][row] * __expf(sM[w][row] - m);
        m_fin[r] = m;
        l_fin[r] = 1.0f / l;
    }

    // pass 2: recompute S, write normalized P (bf16)
    for (int jt = j0; jt < j0 + 512; jt += 16) {
        v8f s = s_tile(sQ, Kb, jt, lrow, hi);
        #pragma unroll
        for (int r = 0; r < 8; ++r) {
            float p = __expf(s[r] * scale - m_fin[r]) * l_fin[r];
            int row = ibase + hi * 8 + r;
            Pb[(size_t)row * 4096 + jt + lrow] = f2bf(p);
        }
    }
}

// ---------------------------------------------------------------- launch
extern "C" void kernel_launch(void* const* d_in, const int* in_sizes, int n_in,
                              void* d_out, int out_size, void* d_ws, size_t ws_size,
                              hipStream_t stream) {
    const float* x  = (const float*)d_in[0];
    const float* Wq = (const float*)d_in[1];
    const float* Wk = (const float*)d_in[2];
    const float* Wv = (const float*)d_in[3];

    char* ws = (char*)d_ws;
    unsigned short* xb  = (unsigned short*)(ws);                 //  8 MiB: x  bf16 (4096x1024)
    unsigned short* Wqb = (unsigned short*)(ws + 8388608);       //  2 MiB
    unsigned short* Wkb = Wqb + 1048576;                         //  2 MiB
    unsigned short* Wvb = Wkb + 1048576;                         //  2 MiB
    unsigned short* Qb  = (unsigned short*)(ws + 14680064);      //  8 MiB: Q  bf16 (4096x1024)
    unsigned short* Kb  = (unsigned short*)(ws + 23068672);      //  8 MiB: K  bf16 (4096x1024)
    unsigned short* Vtb = (unsigned short*)(ws + 31457280);      //  8 MiB: V^T bf16 (1024x4096)
    unsigned short* Pb  = (unsigned short*)(ws + 39845888);      // 32 MiB: P  bf16 (4096x4096)
    float* out = (float*)d_out;

    cvt_f32_bf16<<<1024, 256, 0, stream>>>(x,  xb,  4096 * 1024);
    cvt_f32_bf16<<<512,  256, 0, stream>>>(Wq, Wqb, 1024 * 1024);
    cvt_f32_bf16<<<512,  256, 0, stream>>>(Wk, Wkb, 1024 * 1024);
    cvt_f32_bf16<<<512,  256, 0, stream>>>(Wv, Wvb, 1024 * 1024);

    dim3 gqkv(4096 / BM, 1024 / BN);
    gemm_bf16<<<gqkv, 128, 0, stream>>>(xb, 1024, Wqb, 1024, Qb,  1024, 0, 1024);
    gemm_bf16<<<gqkv, 128, 0, stream>>>(xb, 1024, Wkb, 1024, Kb,  1024, 0, 1024);
    gemm_bf16<<<gqkv, 128, 0, stream>>>(xb, 1024, Wvb, 1024, Vtb, 1024, 1, 4096);

    attn_softmax<<<256, 256, 0, stream>>>(Qb, Kb, Pb);

    dim3 gout(4096 / BM, 1024 / BN);
    gemm_bf16<<<gout, 128, 0, stream>>>(Pb, 4096, Vtb, 4096, out, 4096, 2, 1024);
}
